// SwinTransformer_28149215658414
// MI455X (gfx1250) — matrix operand order repaired and checked by reading, not compiled, not run on hardware
//
#include <hip/hip_runtime.h>
#include <hip/hip_fp16.h>

typedef __attribute__((ext_vector_type(16))) _Float16 v16h;
typedef __attribute__((ext_vector_type(8)))  float    v8f;

#define WS_    7
#define SH_    3
#define NH_    8
#define NTOK   49
#define NP     64       // padded tokens per window
#define C_     256
#define HD_    32
#define IMG_   56
#define NWTOT  2048     // 32 images * 64 windows
#define QSCALE 0.17677669529663687f   // 32^-0.5
#define LDA    264      // LDS row stride (halfs) for 256-wide tiles
#define LDS_S  65       // LDS row stride (floats) for S
#define LDS_P  72       // LDS row stride (halfs) for P

// ---- WMMA helpers -----------------------------------------------------------

__device__ __forceinline__ v8f wmma16(v16h a, v16h b, v8f c) {
  // D = A(16x32 f16) * B(32x16 f16) + C(16x16 f32)
  return __builtin_amdgcn_wmma_f32_16x16x32_f16(false, a, false, b,
                                                (short)0, c, false, false);
}

// A-fragment (16x32, f16), row-major source, row stride rs (halfs).
// Lane L: m = L&15; k(e) = (e<8) ? g*8+e : 16+g*8+(e-8), g = L>>4.
__device__ __forceinline__ v16h load_a_frag(const _Float16* base, int rs) {
  int lane = threadIdx.x & 31;
  int m = lane & 15, g = lane >> 4;
  const _Float16* p0 = base + m * rs + g * 8;
  const _Float16* p1 = base + m * rs + 16 + g * 8;
  v16h a;
#pragma unroll
  for (int i = 0; i < 8; i++) { a[i] = p0[i]; a[i + 8] = p1[i]; }
  return a;
}

// B-fragment (32x16, f16) where column n of B is row n of `base`.
// Lane L: n = L&15; k(e) = (e<8) ? g*8+e : 16+g*8+(e-8), g = L>>4.
__device__ __forceinline__ v16h load_b_frag(const _Float16* base, int rs) {
  int lane = threadIdx.x & 31;
  int n = lane & 15, g = lane >> 4;
  const _Float16* p0 = base + n * rs + g * 8;
  const _Float16* p1 = base + n * rs + 16 + g * 8;
  v16h b;
#pragma unroll
  for (int i = 0; i < 8; i++) { b[i] = p0[i]; b[i + 8] = p1[i]; }
  return b;
}

// ---- Kernel 1: fp32 -> fp16 weight convert ----------------------------------

__global__ void swin_cvt_f16(const float* __restrict__ src,
                             _Float16* __restrict__ dst, int n) {
  int i = blockIdx.x * 256 + threadIdx.x;
  if (i < n) dst[i] = (_Float16)src[i];
}

// ---- Kernel 2: roll(-3,-3) + window partition + f16, zero-pad tokens --------

__global__ void swin_partition(const float* __restrict__ x,
                               _Float16* __restrict__ Xw) {
  int w = blockIdx.x;          // window 0..2047
  int t = blockIdx.y;          // padded token 0..63
  int c = threadIdx.x;         // channel 0..255
  int b = w >> 6, wid = w & 63, wy = wid >> 3, wx = wid & 7;
  _Float16 v = (_Float16)0.f;
  if (t < NTOK) {
    int ti = t / 7, tj = t - (t / 7) * 7;
    int sh = (wy * 7 + ti + SH_) % IMG_;
    int sw = (wx * 7 + tj + SH_) % IMG_;
    v = (_Float16)x[(((size_t)b * IMG_ + sh) * IMG_ + sw) * C_ + c];
  }
  Xw[((size_t)w * NP + t) * C_ + c] = v;
}

// ---- Kernel 3: QKV GEMM (per-window, LDS-tiled, 8 waves x 6 tiles) ---------

__global__ __launch_bounds__(256) void swin_qkv(
    const _Float16* __restrict__ Xw, const _Float16* __restrict__ Wq,
    const float* __restrict__ qbias, _Float16* __restrict__ Qb,
    _Float16* __restrict__ Kb, _Float16* __restrict__ Vt) {
  __shared__ _Float16 sA[NP * LDA];
  int w = blockIdx.x, tid = threadIdx.x;
  // cooperative 64x256-half stage (2048 x 16B chunks)
  const uint4* gsrc = (const uint4*)(Xw + (size_t)w * NP * C_);
#pragma unroll
  for (int i = 0; i < 8; i++) {
    int id = tid + i * 256;
    int row = id >> 5, col = id & 31;
    *(uint4*)(&sA[row * LDA + col * 8]) = gsrc[id];
  }
  __syncthreads();
  int wv = tid >> 5, lane = tid & 31, g = lane >> 4, ln = lane & 15;
  for (int tt = 0; tt < 6; tt++) {
    int ct = wv * 6 + tt;           // output 16-col tile, 0..47 (768 cols)
    int c0 = ct * 16;
    v8f acc[4] = {};
    for (int ks = 0; ks < 8; ks++) {          // K = 256 = 8 x 32
      v16h bf = load_b_frag(Wq + (size_t)c0 * C_ + ks * 32, C_);
#pragma unroll
      for (int mt = 0; mt < 4; mt++) {
        v16h af = load_a_frag(&sA[mt * 16 * LDA + ks * 32], LDA);
        acc[mt] = wmma16(af, bf, acc[mt]);
      }
    }
    int c = c0 + ln;                // 0..767
    int which = c >> 8;             // 0=q 1=k 2=v
    int h = (c >> 5) & 7;
    int hdl = c & 31;
    float bs = qbias[c];
    size_t base = ((size_t)w * NH_ + h) * NP * HD_;
#pragma unroll
    for (int mt = 0; mt < 4; mt++) {
#pragma unroll
      for (int r = 0; r < 8; r++) {
        int m = mt * 16 + g * 8 + r;
        float val = (m < NTOK) ? (acc[mt][r] + bs) : 0.f;   // zero-pad tokens
        if (which == 0)      Qb[base + m * HD_ + hdl] = (_Float16)(val * QSCALE);
        else if (which == 1) Kb[base + m * HD_ + hdl] = (_Float16)val;
        else                 Vt[((size_t)w * NH_ + h) * HD_ * NP + hdl * NP + m]
                               = (_Float16)val;             // V stored transposed
      }
    }
  }
}

// ---- Kernel 4: attention, one wave per (window, head) -----------------------

__global__ __launch_bounds__(32) void swin_attn(
    const _Float16* __restrict__ Qb, const _Float16* __restrict__ Kb,
    const _Float16* __restrict__ Vt, const float* __restrict__ rpb,
    _Float16* __restrict__ Xo) {
  __shared__ float    S[NP * LDS_S];
  __shared__ _Float16 P[NP * LDS_P];
  int wh = blockIdx.x;                 // w*8 + h
  int w = wh >> 3, h = wh & 7;
  int wid = w & 63, wy = wid >> 3, wx = wid & 7;
  int lane = threadIdx.x, g = lane >> 4, ln = lane & 15;
  const _Float16* Qp = Qb + (size_t)wh * NP * HD_;
  const _Float16* Kp = Kb + (size_t)wh * NP * HD_;
  const _Float16* Vp = Vt + (size_t)wh * HD_ * NP;

  v16h qa[4], kf[4];                   // hd=32 == one WMMA K step
#pragma unroll
  for (int t = 0; t < 4; t++) {
    qa[t] = load_a_frag(Qp + t * 16 * HD_, HD_);
    kf[t] = load_b_frag(Kp + t * 16 * HD_, HD_);
  }
  // S = Q K^T + rel-pos bias + shift mask  (16 WMMA)
#pragma unroll
  for (int mt = 0; mt < 4; mt++) {
#pragma unroll
    for (int nt = 0; nt < 4; nt++) {
      v8f acc = {};
      acc = wmma16(qa[mt], kf[nt], acc);
#pragma unroll
      for (int r = 0; r < 8; r++) {
        int m = mt * 16 + g * 8 + r;
        int n = nt * 16 + ln;
        float sv;
        if (n >= NTOK)      sv = -1e30f;      // padded key -> never attended
        else if (m >= NTOK) sv = 0.f;         // padded query row, ignored later
        else {
          int mi = m / 7, mj = m - mi * 7;
          int ni = n / 7, nj = n - ni * 7;
          int idx = (mi - ni + 6) * 13 + (mj - nj + 6);
          float bsv = rpb[idx * NH_ + h];
          int phm = wy * 7 + mi, pwm = wx * 7 + mj;
          int phn = wy * 7 + ni, pwn = wx * 7 + nj;
          int rm = (phm < 49 ? 0 : (phm < 53 ? 1 : 2)) * 3 +
                   (pwm < 49 ? 0 : (pwm < 53 ? 1 : 2));
          int rn = (phn < 49 ? 0 : (phn < 53 ? 1 : 2)) * 3 +
                   (pwn < 49 ? 0 : (pwn < 53 ? 1 : 2));
          sv = acc[r] + bsv + ((rm != rn) ? -100.f : 0.f);
        }
        S[m * LDS_S + n] = sv;
      }
    }
  }
  __syncthreads();
  // row softmax, lane handles rows {lane, lane+32}; stride-65 -> conflict-free
#pragma unroll
  for (int rr = 0; rr < 2; rr++) {
    int row = lane + rr * 32;
    float mx = -1e30f;
    for (int j = 0; j < NP; j++) mx = fmaxf(mx, S[row * LDS_S + j]);
    float sum = 0.f;
    for (int j = 0; j < NP; j++) {
      float e = __expf(S[row * LDS_S + j] - mx);
      sum += e;
      S[row * LDS_S + j] = e;
    }
    float inv = 1.f / sum;
    for (int j = 0; j < NP; j++)
      P[row * LDS_P + j] = (_Float16)(S[row * LDS_S + j] * inv);
  }
  __syncthreads();
  // O = P V  (16 WMMA), scatter heads into Xo[w][token][C]
#pragma unroll
  for (int mt = 0; mt < 4; mt++) {
#pragma unroll
    for (int nt = 0; nt < 2; nt++) {
      v8f acc = {};
#pragma unroll
      for (int kt = 0; kt < 2; kt++) {
        v16h pa = load_a_frag(&P[mt * 16 * LDS_P + kt * 32], LDS_P);
        v16h vb = load_b_frag(Vp + (nt * 16) * NP + kt * 32, NP);
        acc = wmma16(pa, vb, acc);
      }
#pragma unroll
      for (int r = 0; r < 8; r++) {
        int m = mt * 16 + g * 8 + r;
        int n = nt * 16 + ln;
        Xo[((size_t)w * NP + m) * C_ + h * HD_ + n] = (_Float16)acc[r];
      }
    }
  }
}

// ---- Kernel 5: proj GEMM + bias + window-reverse + roll(+3,+3) --------------

__global__ __launch_bounds__(256) void swin_proj(
    const _Float16* __restrict__ Xo, const _Float16* __restrict__ Wp,
    const float* __restrict__ pbias, float* __restrict__ out) {
  __shared__ _Float16 sA[NP * LDA];
  int w = blockIdx.x, tid = threadIdx.x;
  const uint4* gsrc = (const uint4*)(Xo + (size_t)w * NP * C_);
#pragma unroll
  for (int i = 0; i < 8; i++) {
    int id = tid + i * 256;
    int row = id >> 5, col = id & 31;
    *(uint4*)(&sA[row * LDA + col * 8]) = gsrc[id];
  }
  __syncthreads();
  int wv = tid >> 5, lane = tid & 31, g = lane >> 4, ln = lane & 15;
  int b = w >> 6, wid = w & 63, wy = wid >> 3, wx = wid & 7;
  for (int tt = 0; tt < 2; tt++) {
    int ct = wv * 2 + tt;           // 16 output tiles (256 cols) / 8 waves
    int c0 = ct * 16;
    v8f acc[4] = {};
    for (int ks = 0; ks < 8; ks++) {
      v16h bf = load_b_frag(Wp + (size_t)c0 * C_ + ks * 32, C_);
#pragma unroll
      for (int mt = 0; mt < 4; mt++) {
        v16h af = load_a_frag(&sA[mt * 16 * LDA + ks * 32], LDA);
        acc[mt] = wmma16(af, bf, acc[mt]);
      }
    }
    int c = c0 + ln;
    float bs = pbias[c];
#pragma unroll
    for (int mt = 0; mt < 4; mt++) {
#pragma unroll
      for (int r = 0; r < 8; r++) {
        int m = mt * 16 + g * 8 + r;
        if (m < NTOK) {
          int ti = m / 7, tj = m - ti * 7;
          int oh = (wy * 7 + ti + SH_) % IMG_;
          int ow = (wx * 7 + tj + SH_) % IMG_;
          out[(((size_t)b * IMG_ + oh) * IMG_ + ow) * C_ + c] = acc[mt][r] + bs;
        }
      }
    }
  }
}

// ---- launcher ---------------------------------------------------------------

extern "C" void kernel_launch(void* const* d_in, const int* in_sizes, int n_in,
                              void* d_out, int out_size, void* d_ws, size_t ws_size,
                              hipStream_t stream) {
  (void)in_sizes; (void)n_in; (void)out_size; (void)ws_size;
  const float* x     = (const float*)d_in[0];
  const float* qkvw  = (const float*)d_in[1];
  const float* qkvb  = (const float*)d_in[2];
  const float* projw = (const float*)d_in[3];
  const float* projb = (const float*)d_in[4];
  const float* rpb   = (const float*)d_in[5];

  char* ws = (char*)d_ws;
  size_t off = 0;
  auto carve = [&](size_t bytes) -> char* {
    char* p = ws + off;
    off += (bytes + 255) & ~(size_t)255;
    return p;
  };
  _Float16* Xw   = (_Float16*)carve((size_t)NWTOT * NP * C_ * 2);          // 64 MB
  _Float16* Qb   = (_Float16*)carve((size_t)NWTOT * NH_ * NP * HD_ * 2);   // 64 MB
  _Float16* Kb   = (_Float16*)carve((size_t)NWTOT * NH_ * NP * HD_ * 2);   // 64 MB
  _Float16* Vt   = (_Float16*)carve((size_t)NWTOT * NH_ * HD_ * NP * 2);   // 64 MB
  _Float16* Wq16 = (_Float16*)carve((size_t)768 * C_ * 2);
  _Float16* Wp16 = (_Float16*)carve((size_t)C_ * C_ * 2);
  _Float16* Xo   = Xw;   // Xw fully consumed by swin_qkv before swin_attn writes

  swin_cvt_f16<<<(768 * C_ + 255) / 256, 256, 0, stream>>>(qkvw, Wq16, 768 * C_);
  swin_cvt_f16<<<(C_ * C_ + 255) / 256, 256, 0, stream>>>(projw, Wp16, C_ * C_);
  swin_partition<<<dim3(NWTOT, NP), 256, 0, stream>>>(x, Xw);
  swin_qkv<<<NWTOT, 256, 0, stream>>>(Xw, Wq16, qkvb, Qb, Kb, Vt);
  swin_attn<<<NWTOT * NH_, 32, 0, stream>>>(Qb, Kb, Vt, rpb, Xo);
  swin_proj<<<NWTOT, 256, 0, stream>>>(Xo, Wp16, projb, (float*)d_out);
}
